// CYActorNetwork_71476845740358
// MI455X (gfx1250) — matrix operand order repaired and checked by reading, not compiled, not run on hardware
//
#include <hip/hip_runtime.h>
#include <hip/hip_bf16.h>
#include <math.h>

// ---------------------------------------------------------------------------
// Problem constants (match reference)
// ---------------------------------------------------------------------------
#define NN   4096      // nodes
#define EE   131072    // edges
#define DIN  16
#define DD   256       // hidden dim
#define AA   64        // action dim
#define HH   4         // heads
#define DH   64        // head dim

typedef __attribute__((ext_vector_type(16))) _Float16 v16h;
typedef __attribute__((ext_vector_type(8)))  float    v8f;
typedef __attribute__((ext_vector_type(4)))  float    f32x4;
typedef __attribute__((ext_vector_type(4)))  _Float16 f16x4;

__device__ __forceinline__ v8f wmma16(v16h a, v16h b, v8f c) {
    // D = A(16x32 f16) * B(32x16 f16) + C(16x16 f32)
    return __builtin_amdgcn_wmma_f32_16x16x32_f16(
        /*neg_a=*/false, a, /*neg_b=*/false, b,
        /*c_mod=*/(short)0, c, /*reuse_a=*/false, /*reuse_b=*/false);
}

__device__ __forceinline__ float gelu_exact(float x) {
    return 0.5f * x * (1.0f + erff(x * 0.7071067811865475f));
}

// ---------------------------------------------------------------------------
// Intra-16-lane xor-butterfly reductions. Prefer v_permlane16_b32 (pure VALU,
// co-issues with WMMA) over ds_bpermute chains.
// ---------------------------------------------------------------------------
#if __has_builtin(__builtin_amdgcn_permlane16)
#define PERMLANE16(v, s0, s1) __builtin_amdgcn_permlane16((v), (v), (s0), (s1), false, false)
#elif __has_builtin(__builtin_amdgcn_permlane16_b32)
#define PERMLANE16(v, s0, s1) __builtin_amdgcn_permlane16_b32((v), (v), (s0), (s1), false, false)
#endif

__device__ __forceinline__ float xl16(float v, unsigned s0, unsigned s1, int xorm) {
#ifdef PERMLANE16
    (void)xorm;
    int r = PERMLANE16(__float_as_int(v), (int)s0, (int)s1);
    return __int_as_float(r);
#else
    (void)s0; (void)s1;
    return __shfl_xor(v, xorm, 32);
#endif
}
__device__ __forceinline__ float red16_max(float v) {
    v = fmaxf(v, xl16(v, 0x67452301u, 0xEFCDAB89u, 1));   // lane ^ 1
    v = fmaxf(v, xl16(v, 0x54761032u, 0xDCFE98BAu, 2));   // lane ^ 2
    v = fmaxf(v, xl16(v, 0x32107654u, 0xBA98FEDCu, 4));   // lane ^ 4
    v = fmaxf(v, xl16(v, 0xFEDCBA98u, 0x76543210u, 8));   // lane ^ 8
    return v;
}
__device__ __forceinline__ float red16_sum(float v) {
    v += xl16(v, 0x67452301u, 0xEFCDAB89u, 1);
    v += xl16(v, 0x54761032u, 0xDCFE98BAu, 2);
    v += xl16(v, 0x32107654u, 0xBA98FEDCu, 4);
    v += xl16(v, 0xFEDCBA98u, 0x76543210u, 8);
    return v;
}

// ---------------------------------------------------------------------------
// Degree / symmetric normalization
// ---------------------------------------------------------------------------
__global__ void k_deg_init(float* deg) {
    int n = blockIdx.x * blockDim.x + threadIdx.x;
    if (n < NN) deg[n] = 1.0f;                      // self-loop
}
__global__ void k_deg_scatter(const int* __restrict__ ei, float* deg) {
    int e = blockIdx.x * blockDim.x + threadIdx.x;
    if (e < EE) atomicAdd(&deg[ei[EE + e]], 1.0f);  // dst = ei[1][e]
}
__global__ void k_dinv(float* deg) {
    int n = blockIdx.x * blockDim.x + threadIdx.x;
    if (n < NN) deg[n] = rsqrtf(deg[n]);            // deg >= 1 always
}

// ---------------------------------------------------------------------------
// WMMA GEMM: C[M,Nn] = A[M,K] @ B  (+bias)
//   BT=0 : B stored [K,Nn] row-major
//   BT=1 : B stored [Nn,K] row-major (torch weight, computes A @ B^T)
// 128 threads (4 waves). Tile 64(M) x 64(N). K in 32-chunks, zero-padded.
// LDS: As[m][k], Bs[n][k] (both k-contiguous -> b64/b128-mergeable reads).
// ---------------------------------------------------------------------------
template <bool BT, bool BIAS>
__global__ void k_gemm(const float* __restrict__ A, const float* __restrict__ B,
                       const float* __restrict__ bias, float* __restrict__ C,
                       int M, int K, int Nn) {
    __shared__ alignas(16) _Float16 As[64][40];
    __shared__ alignas(16) _Float16 Bs[64][40];

    const int tid  = threadIdx.x;
    const int wid  = tid >> 5;
    const int lane = tid & 31;
    const int half = lane >> 4;
    const int lm   = lane & 15;
    const int mbase = blockIdx.x * 64;
    const int nbase = blockIdx.y * 64;
    const int kb8  = half * 8;
    const int kb16 = half * 16;

    v8f acc[4] = {};

    const int ktiles = (K + 31) >> 5;
    for (int kt = 0; kt < ktiles; ++kt) {
        const int kb = kt * 32;
        __syncthreads();
        // stage A: 64 rows x 32 k  (512 float4 groups)
        #pragma unroll
        for (int i = 0; i < 4; ++i) {
            int idx = i * 128 + tid;
            int r = idx >> 3, c4 = (idx & 7) * 4;
            f32x4 v = {};
            if (kb + c4 < K) v = *(const f32x4*)&A[(size_t)(mbase + r) * K + kb + c4];
            f16x4 hv = {(_Float16)v[0], (_Float16)v[1], (_Float16)v[2], (_Float16)v[3]};
            *(f16x4*)&As[r][c4] = hv;
        }
        // stage B -> Bs[n][k]
        if (BT) {
            #pragma unroll
            for (int i = 0; i < 4; ++i) {
                int idx = i * 128 + tid;
                int n = idx >> 3, k4 = (idx & 7) * 4;
                f32x4 v = {};
                if (kb + k4 < K) v = *(const f32x4*)&B[(size_t)(nbase + n) * K + kb + k4];
                f16x4 hv = {(_Float16)v[0], (_Float16)v[1], (_Float16)v[2], (_Float16)v[3]};
                *(f16x4*)&Bs[n][k4] = hv;
            }
        } else {
            #pragma unroll
            for (int i = 0; i < 4; ++i) {
                int idx = i * 128 + tid;
                int r = idx >> 4, c4 = (idx & 15) * 4;   // r = k row, c = n
                f32x4 v = {};
                if (kb + r < K) v = *(const f32x4*)&B[(size_t)(kb + r) * Nn + nbase + c4];
                Bs[c4 + 0][r] = (_Float16)v[0];
                Bs[c4 + 1][r] = (_Float16)v[1];
                Bs[c4 + 2][r] = (_Float16)v[2];
                Bs[c4 + 3][r] = (_Float16)v[3];
            }
        }
        __syncthreads();

        // A fragment: rows wid*16 .. wid*16+15 (ISA A layout)
        v16h af;
        #pragma unroll
        for (int j = 0; j < 16; ++j) {
            int k = kb8 + j + ((j >= 8) ? 8 : 0);
            af[j] = As[wid * 16 + lm][k];
        }
        #pragma unroll
        for (int nc = 0; nc < 4; ++nc) {
            v16h bf;
            #pragma unroll
            for (int j = 0; j < 16; ++j) bf[j] = Bs[nc * 16 + lm][kb8 + j + ((j >= 8) ? 8 : 0)];
            acc[nc] = wmma16(af, bf, acc[nc]);
        }
    }

    #pragma unroll
    for (int nc = 0; nc < 4; ++nc) {
        #pragma unroll
        for (int i = 0; i < 8; ++i) {
            int row = mbase + wid * 16 + i + half * 8;
            int col = nbase + nc * 16 + lm;
            float v = acc[nc][i];
            if (BIAS) v += bias[col];
            C[(size_t)row * Nn + col] = v;
        }
    }
}

// ---------------------------------------------------------------------------
// GCN aggregation: out = scatter_add(h[src]*dinv[src]*dinv[dst]) + selfloop + b
// ---------------------------------------------------------------------------
__global__ void k_agg_self(const float* __restrict__ h, const float* __restrict__ dinv,
                           const float* __restrict__ b, float* __restrict__ out) {
    int n = blockIdx.x, d = threadIdx.x;
    float di = dinv[n];
    out[(size_t)n * DD + d] = h[(size_t)n * DD + d] * di * di + b[d];
}
__global__ void k_agg_edge(const float* __restrict__ h, const float* __restrict__ dinv,
                           const int* __restrict__ ei, float* __restrict__ out) {
    int e = blockIdx.x, d = threadIdx.x;
    int s = ei[e], t = ei[EE + e];
    float nrm = dinv[s] * dinv[t];
    atomicAdd(&out[(size_t)t * DD + d], h[(size_t)s * DD + d] * nrm);
}
__global__ void k_gelu(float* __restrict__ x, int n) {
    int i = blockIdx.x * blockDim.x + threadIdx.x;
    if (i < n) x[i] = gelu_exact(x[i]);
}

// ---------------------------------------------------------------------------
// Flash attention. qkv layout [N, 3*DD]: q at +0, k at +DD, v at +2*DD,
// head h occupies cols [h*DH, h*DH+DH). One WG per (64-query block, head);
// 4 waves, each wave owns 16 query rows. 32-key tiles, online softmax.
// ---------------------------------------------------------------------------
__global__ void k_attn(const float* __restrict__ qkv, float* __restrict__ out) {
    __shared__ alignas(16) _Float16 Kt[32][72];
    __shared__ alignas(16) _Float16 Vt[32][72];
    __shared__ alignas(16) _Float16 Pt[4][16][36];

    const int tid  = threadIdx.x;
    const int wid  = tid >> 5;
    const int lane = tid & 31;
    const int half = lane >> 4;
    const int lm   = lane & 15;
    const int head  = blockIdx.y;
    const int qbase = blockIdx.x * 64 + wid * 16;
    const int hoff  = head * DH;
    const int kb8   = half * 8;
    const int kb16  = half * 16;

    // Q fragments (A layout), pre-scaled by 1/sqrt(DH)
    v16h qf[2];
    #pragma unroll
    for (int kk = 0; kk < 2; ++kk) {
        #pragma unroll
        for (int j = 0; j < 16; ++j) {
            int k = kk * 32 + kb8 + j + ((j >= 8) ? 8 : 0);
            qf[kk][j] = (_Float16)(qkv[(size_t)(qbase + lm) * (3 * DD) + hoff + k] * 0.125f);
        }
    }

    v8f o[4] = {};
    float mrow[8], lrow[8];
    #pragma unroll
    for (int i = 0; i < 8; ++i) { mrow[i] = -1e30f; lrow[i] = 0.0f; }

    // staging indices for this thread (512 float4 groups per 32x64 tile)
    const int sr = tid >> 4;            // used with +32*i rows? no: idx mapping below
    (void)sr;

    for (int t = 0; t < NN / 32; ++t) {
        __syncthreads();
        // stage K,V tiles (32 keys x 64 dims) as float4 -> packed f16x4
        #pragma unroll
        for (int i = 0; i < 4; ++i) {
            int idx = i * 128 + tid;
            int r = idx >> 4, c4 = (idx & 15) * 4;
            size_t base = (size_t)(t * 32 + r) * (3 * DD) + hoff + c4;
            f32x4 kv = *(const f32x4*)&qkv[base + DD];
            f32x4 vv = *(const f32x4*)&qkv[base + 2 * DD];
            f16x4 kh = {(_Float16)kv[0], (_Float16)kv[1], (_Float16)kv[2], (_Float16)kv[3]};
            f16x4 vh = {(_Float16)vv[0], (_Float16)vv[1], (_Float16)vv[2], (_Float16)vv[3]};
            *(f16x4*)&Kt[r][c4] = kh;
            *(f16x4*)&Vt[r][c4] = vh;
            if (t + 1 < NN / 32)
                __builtin_prefetch(&qkv[base + 32 * (3 * DD) + DD], 0, 3);
        }
        __syncthreads();

        // S = Q @ K^T  (16 x 32, two 16x16 fragments)
        v8f s0 = {}, s1 = {};
        #pragma unroll
        for (int ks = 0; ks < 2; ++ks) {
            v16h bk0, bk1;
            #pragma unroll
            for (int j = 0; j < 16; ++j) {
                int k = ks * 32 + kb8 + j + ((j >= 8) ? 8 : 0);
                bk0[j] = Kt[lm][k];          // keys 0..15
                bk1[j] = Kt[16 + lm][k];     // keys 16..31
            }
            s0 = wmma16(qf[ks], bk0, s0);
            s1 = wmma16(qf[ks], bk1, s1);
        }

        // online softmax (row stats per C-fragment element; VALU lane reductions)
        #pragma unroll
        for (int i = 0; i < 8; ++i) {
            float mx = red16_max(fmaxf(s0[i], s1[i]));
            float mnew  = fmaxf(mrow[i], mx);
            float alpha = __expf(mrow[i] - mnew);
            float p0 = __expf(s0[i] - mnew);
            float p1 = __expf(s1[i] - mnew);
            float ps = red16_sum(p0 + p1);
            lrow[i] = lrow[i] * alpha + ps;
            mrow[i] = mnew;
            #pragma unroll
            for (int nc = 0; nc < 4; ++nc) o[nc][i] *= alpha;
            Pt[wid][i + half * 8][lm]      = (_Float16)p0;
            Pt[wid][i + half * 8][16 + lm] = (_Float16)p1;
        }
        __syncthreads();

        // P fragment (A layout) from LDS
        v16h pf;
        #pragma unroll
        for (int j = 0; j < 16; ++j) {
            int k = kb8 + j + ((j >= 8) ? 8 : 0);
            pf[j] = Pt[wid][lm][k];
        }
        // O += P @ V
        #pragma unroll
        for (int nc = 0; nc < 4; ++nc) {
            v16h bv;
            #pragma unroll
            for (int j = 0; j < 16; ++j) bv[j] = Vt[kb8 + j + ((j >= 8) ? 8 : 0)][nc * 16 + lm];
            o[nc] = wmma16(pf, bv, o[nc]);
        }
    }

    // normalize and write [N, DD] slice for this head
    #pragma unroll
    for (int i = 0; i < 8; ++i) {
        float rl = 1.0f / lrow[i];
        int row = qbase + i + half * 8;
        #pragma unroll
        for (int nc = 0; nc < 4; ++nc) {
            int col = hoff + nc * 16 + lm;
            out[(size_t)row * DD + col] = o[nc][i] * rl;
        }
    }
}

// ---------------------------------------------------------------------------
// Mean pool over nodes -> h_global (two-stage)
// ---------------------------------------------------------------------------
__global__ void k_colsum_partial(const float* __restrict__ att, float* __restrict__ part) {
    int b = blockIdx.x, d = threadIdx.x;     // 32 blocks x 128 rows
    float s = 0.0f;
    for (int n = b * 128; n < b * 128 + 128; ++n) s += att[(size_t)n * DD + d];
    part[(size_t)b * DD + d] = s;
}
__global__ void k_colsum_final(const float* __restrict__ part, float* __restrict__ hg,
                               float* __restrict__ out_hg) {
    int d = threadIdx.x;
    float s = 0.0f;
    for (int b = 0; b < 32; ++b) s += part[(size_t)b * DD + d];
    s *= (1.0f / NN);
    hg[d] = s;
    out_hg[d] = s;
}

// ---------------------------------------------------------------------------
// Head: Linear(256->512) -> LayerNorm -> GELU -> Linear(512->64) -> expmap0
// ---------------------------------------------------------------------------
__global__ void k_head(const float* __restrict__ hg,
                       const float* __restrict__ Wd, const float* __restrict__ bd,
                       const float* __restrict__ lng, const float* __restrict__ lnb,
                       const float* __restrict__ Wh, const float* __restrict__ bh,
                       float* __restrict__ out) {
    __shared__ float red[512];
    __shared__ float tsh[512];
    __shared__ float hgs[DD];
    const int tid = threadIdx.x;

    if (tid < DD) hgs[tid] = hg[tid];
    __syncthreads();

    float acc = bd[tid];
    for (int k = 0; k < DD; ++k) acc += hgs[k] * Wd[(size_t)k * 512 + tid];

    // layernorm over 512
    red[tid] = acc; __syncthreads();
    for (int s = 256; s > 0; s >>= 1) { if (tid < s) red[tid] += red[tid + s]; __syncthreads(); }
    float mu = red[0] * (1.0f / 512.0f); __syncthreads();
    float dx = acc - mu;
    red[tid] = dx * dx; __syncthreads();
    for (int s = 256; s > 0; s >>= 1) { if (tid < s) red[tid] += red[tid + s]; __syncthreads(); }
    float var = red[0] * (1.0f / 512.0f); __syncthreads();
    float y = dx * rsqrtf(var + 1e-5f) * lng[tid] + lnb[tid];
    tsh[tid] = gelu_exact(y);
    __syncthreads();

    // Linear 512 -> 64
    float v = 0.0f;
    if (tid < AA) {
        v = bh[tid];
        for (int k = 0; k < 512; ++k) v += tsh[k] * Wh[(size_t)k * AA + tid];
    }
    red[tid] = (tid < AA) ? v * v : 0.0f; __syncthreads();
    for (int s = 256; s > 0; s >>= 1) { if (tid < s) red[tid] += red[tid + s]; __syncthreads(); }
    float nrm = fmaxf(sqrtf(red[0]), 1e-5f);
    if (tid < AA) out[tid] = tanhf(nrm) * v / nrm;   // expmap0, c = 1
}

// ---------------------------------------------------------------------------
// Launch
// ---------------------------------------------------------------------------
extern "C" void kernel_launch(void* const* d_in, const int* in_sizes, int n_in,
                              void* d_out, int out_size, void* d_ws, size_t ws_size,
                              hipStream_t stream) {
    const float* x   = (const float*)d_in[0];
    const int*   ei  = (const int*)d_in[1];
    const float* W1  = (const float*)d_in[2];
    const float* b1  = (const float*)d_in[3];
    const float* W2  = (const float*)d_in[4];
    const float* b2  = (const float*)d_in[5];
    const float* W3  = (const float*)d_in[6];
    const float* b3  = (const float*)d_in[7];
    const float* inw = (const float*)d_in[8];
    const float* inb = (const float*)d_in[9];
    const float* outw= (const float*)d_in[10];
    const float* outb= (const float*)d_in[11];
    const float* Wd  = (const float*)d_in[12];
    const float* bd  = (const float*)d_in[13];
    const float* lng = (const float*)d_in[14];
    const float* lnb = (const float*)d_in[15];
    const float* Wh  = (const float*)d_in[16];
    const float* bh  = (const float*)d_in[17];
    float* outp = (float*)d_out;      // [0:64) deformation, [64:320) h_global

    float* ws   = (float*)d_ws;
    float* dinv = ws;                         // N
    float* G    = dinv + NN;                  // N*D
    float* O    = G + (size_t)NN * DD;        // N*D
    float* QKV  = O + (size_t)NN * DD;        // N*3D
    float* HG   = QKV + (size_t)NN * 3 * DD;  // D
    float* PART = HG + DD;                    // 32*D

    // normalization
    k_deg_init<<<NN / 256, 256, 0, stream>>>(dinv);
    k_deg_scatter<<<EE / 256, 256, 0, stream>>>(ei, dinv);
    k_dinv<<<NN / 256, 256, 0, stream>>>(dinv);

    dim3 blk(128);
    // GCN layer 1 (K = 16, zero-padded)
    k_gemm<false, false><<<dim3(NN / 64, DD / 64), blk, 0, stream>>>(x, W1, nullptr, G, NN, DIN, DD);
    k_agg_self<<<NN, DD, 0, stream>>>(G, dinv, b1, O);
    k_agg_edge<<<EE, DD, 0, stream>>>(G, dinv, ei, O);
    k_gelu<<<(NN * DD) / 256, 256, 0, stream>>>(O, NN * DD);
    // GCN layer 2
    k_gemm<false, false><<<dim3(NN / 64, DD / 64), blk, 0, stream>>>(O, W2, nullptr, G, NN, DD, DD);
    k_agg_self<<<NN, DD, 0, stream>>>(G, dinv, b2, O);
    k_agg_edge<<<EE, DD, 0, stream>>>(G, dinv, ei, O);
    k_gelu<<<(NN * DD) / 256, 256, 0, stream>>>(O, NN * DD);
    // GCN layer 3 (no gelu)
    k_gemm<false, false><<<dim3(NN / 64, DD / 64), blk, 0, stream>>>(O, W3, nullptr, G, NN, DD, DD);
    k_agg_self<<<NN, DD, 0, stream>>>(G, dinv, b3, O);
    k_agg_edge<<<EE, DD, 0, stream>>>(G, dinv, ei, O);

    // QKV projection: O @ in_proj_w^T + in_proj_b  -> [N, 3D]
    k_gemm<true, true><<<dim3(NN / 64, (3 * DD) / 64), blk, 0, stream>>>(O, inw, inb, QKV, NN, DD, 3 * DD);
    // attention -> G [N, D]
    k_attn<<<dim3(NN / 64, HH), blk, 0, stream>>>(QKV, G);
    // out projection: G @ out_proj_w^T + out_proj_b -> O
    k_gemm<true, true><<<dim3(NN / 64, DD / 64), blk, 0, stream>>>(G, outw, outb, O, NN, DD, DD);

    // mean pool -> HG and d_out[64:320)
    k_colsum_partial<<<32, DD, 0, stream>>>(O, PART);
    k_colsum_final<<<1, DD, 0, stream>>>(PART, HG, outp + AA);
    // head -> d_out[0:64)
    k_head<<<1, 512, 0, stream>>>(HG, Wd, bd, lng, lnb, Wh, bh, outp);
}
